// CausalConv1dFallback_76330158784656
// MI455X (gfx1250) — compile-verified, matrix operand order checked by reading, not run on hardware
//
#include <hip/hip_runtime.h>

// Causal Conv1d (B=4, C_IN=C_OUT=2048, K=4, S=4096) + bias + SiLU on
// v_wmma_f32_16x16x32_bf16, with double-buffered Tensor-Data-Mover staging.
//
// Path A (ws large enough):
//   1) prepack weights -> bf16, per-tile layout Wp[co][cb][k*32+ci]
//   2) prepack x       -> bf16, transposed + 4 causal zero rows: Xp[b][4+s][ci]
//   3) main kernel: software pipeline -- wave 0 issues tensor_load_to_lds for
//      block cb+1 into the alternate LDS buffer (TDM pad_enable reproduces the
//      bank-conflict-free strides), 8 waves run WMMA on block cb, then wave 0
//      drains TENSORcnt and the workgroup barrier flips buffers.
// Path B (fallback): inline fp32->bf16 conversion staging, no workspace.

constexpr int CBATCH = 4;
constexpr int CIN    = 2048;
constexpr int COUT   = 2048;
constexpr int KW     = 4;
constexpr int SS     = 4096;

constexpr int TILE_CO = 128;
constexpr int TILE_S  = 128;
constexpr int CI_BLK  = 32;
constexpr int NCB     = CIN / CI_BLK;   // 64

constexpr int A_STRIDE = 136;  // ushorts per LDS A row (272B: 256B data + 16B pad)
constexpr int X_STRIDE = 40;   // ushorts per LDS X row (80B: 64B data + 16B pad)
constexpr int X_ROWS   = 132;

constexpr int    XP_ROWS  = SS + 4;                                  // 4100
constexpr size_t WP_BYTES = (size_t)COUT * CIN * KW * 2;             // 33.5 MB
constexpr size_t XP_BYTES = (size_t)CBATCH * XP_ROWS * CIN * 2;      // 67.2 MB

typedef __attribute__((ext_vector_type(16))) __bf16       v16bf;
typedef __attribute__((ext_vector_type(8)))  float        v8f;
typedef __attribute__((ext_vector_type(4)))  unsigned int v4u;
typedef __attribute__((ext_vector_type(4)))  unsigned int u32x4;
typedef __attribute__((ext_vector_type(8)))  int          i32x8;
typedef __attribute__((ext_vector_type(4)))  int          i32x4;

union Frag {
  v16bf bf;
  v4u   u[2];
};

__device__ __forceinline__ unsigned short f2bf(float f) {
  unsigned int u = __builtin_bit_cast(unsigned int, f);
  unsigned int r = u + 0x7FFFu + ((u >> 16) & 1u);
  return (unsigned short)(r >> 16);
}

// ---------------------------------------------------------------------------
// Prepack kernels
// ---------------------------------------------------------------------------

// w[co][ci][k] fp32 -> Wp bf16, ushort layout [(co*NCB+cb)*KW + k][32 ci]
__global__ __launch_bounds__(256)
void prepack_w(const float* __restrict__ w, unsigned short* __restrict__ Wp) {
  int idx = blockIdx.x * 256 + threadIdx.x;   // 2048 co * 1024 ci-pairs
  int co  = idx >> 10;
  int ci  = (idx & 1023) * 2;
  const float* wp0 = w + ((size_t)co * CIN + ci) * KW;
  float4 a = *(const float4*)(wp0);
  float4 b = *(const float4*)(wp0 + 4);
  const float* af = &a.x;
  const float* bf = &b.x;
  int cb  = ci >> 5;
  int cil = ci & 31;
  unsigned int* W32 = (unsigned int*)Wp;
#pragma unroll
  for (int k = 0; k < KW; ++k) {
    unsigned int v = (unsigned int)f2bf(af[k]) | ((unsigned int)f2bf(bf[k]) << 16);
    W32[(((size_t)co * NCB + cb) * KW + k) * 16 + (cil >> 1)] = v;
  }
}

// x[b][ci][s] fp32 -> Xp[b][4+s][ci] bf16 (transpose via LDS)
__global__ __launch_bounds__(256)
void prepack_x(const float* __restrict__ x, unsigned short* __restrict__ Xp) {
  __shared__ unsigned short lT[64 * 32];
  const int tid   = threadIdx.x;
  const int sBlk0 = blockIdx.x * 64;
  const int ci0   = blockIdx.y * 32;
  const int bz    = blockIdx.z;

  int ci_l = tid >> 3;
  int sc   = tid & 7;
  const float* xp = x + ((size_t)(bz * CIN + ci0 + ci_l)) * SS + sBlk0 + sc * 8;
  float4 a = ((const float4*)xp)[0];
  float4 b = ((const float4*)xp)[1];
  const float* af = &a.x;
  const float* bf = &b.x;
#pragma unroll
  for (int j = 0; j < 4; ++j) {
    lT[(sc * 8 + j) * 32 + ci_l]     = f2bf(af[j]);
    lT[(sc * 8 + 4 + j) * 32 + ci_l] = f2bf(bf[j]);
  }
  __syncthreads();

  int row = tid >> 2;
  int q   = tid & 3;
  uint4 v = *(const uint4*)&lT[row * 32 + q * 8];
  uint4* X4 = (uint4*)Xp;
  X4[(((size_t)bz * XP_ROWS + 4 + sBlk0 + row) * CIN + ci0) / 8 + q] = v;
}

// zero the 4 causal pad rows of each batch
__global__ __launch_bounds__(256)
void zero_xpad(unsigned short* __restrict__ Xp) {
  int idx = blockIdx.x * 256 + threadIdx.x;   // 4096 = 4b * 4rows * 256 uint4
  int b   = idx >> 10;
  int row = (idx >> 8) & 3;
  int q   = idx & 255;
  uint4 z = {0u, 0u, 0u, 0u};
  ((uint4*)Xp)[((size_t)b * XP_ROWS + row) * 256 + q] = z;
}

// ---------------------------------------------------------------------------
// TDM descriptor issue: A tile + X tile for ci-block cb into one LDS buffer
// ---------------------------------------------------------------------------
#if defined(__AMDGCN__) && __has_builtin(__builtin_amdgcn_tensor_load_to_lds)
#define HAVE_TDM 1
__device__ __forceinline__ void tdm_issue(const unsigned short* Wp,
                                          const unsigned short* Xp,
                                          int co0, int s0, int bz, int cb,
                                          unsigned ldsA, unsigned ldsX) {
  // A tile: 128 co rows x 128 ushorts, global row stride 8192 elem,
  // LDS pad 16B every 256B -> 272B rows (A_STRIDE)
  unsigned long long gaA = (unsigned long long)(size_t)Wp +
                           ((size_t)co0 * NCB + cb) * 256u;
  u32x4 g0A = {1u, ldsA, (unsigned)gaA,
               (unsigned)((gaA >> 32) & 0x01FFFFFFull) | 0x80000000u};
  const unsigned d0A = (1u << 16) | (1u << 20) | (5u << 22) | (3u << 25);
  i32x8 g1A = {(int)d0A,
               (int)(128u << 16),                 // tensor_dim0 lo
               (int)(128u << 16),                 // tensor_dim0 hi | dim1 lo
               (int)(128u << 16),                 // dim1 hi | tile_dim0
               (int)128,                          // tile_dim1
               (int)8192,                         // tensor_dim0_stride lo
               0, 0};
  // X tile: 132 rows x 32 ushorts, global row stride 2048 elem,
  // LDS pad 16B every 64B -> 80B rows (X_STRIDE)
  unsigned long long gaX = (unsigned long long)(size_t)Xp +
                           (((size_t)bz * XP_ROWS + s0) * CIN + cb * CI_BLK) * 2u;
  u32x4 g0X = {1u, ldsX, (unsigned)gaX,
               (unsigned)((gaX >> 32) & 0x01FFFFFFull) | 0x80000000u};
  const unsigned d0X = (1u << 16) | (1u << 20) | (3u << 22) | (3u << 25);
  i32x8 g1X = {(int)d0X,
               (int)(32u << 16),
               (int)(132u << 16),
               (int)(32u << 16),
               (int)132,
               (int)2048,
               0, 0};
  i32x4 gz = {0, 0, 0, 0};
#if __clang_major__ >= 23
  i32x8 gz8 = {0, 0, 0, 0, 0, 0, 0, 0};
  __builtin_amdgcn_tensor_load_to_lds(g0A, g1A, gz, gz, gz8, 0);
  __builtin_amdgcn_tensor_load_to_lds(g0X, g1X, gz, gz, gz8, 0);
#else
  __builtin_amdgcn_tensor_load_to_lds(g0A, g1A, gz, gz, 0);
  __builtin_amdgcn_tensor_load_to_lds(g0X, g1X, gz, gz, 0);
#endif
}
#else
#define HAVE_TDM 0
#endif

// ---------------------------------------------------------------------------
// Main kernel, packed path: double-buffered TDM pipeline
// ---------------------------------------------------------------------------
__global__ __launch_bounds__(256)
void conv_wmma_packed(const unsigned short* __restrict__ Wp,
                      const unsigned short* __restrict__ Xp,
                      const float* __restrict__ bias,
                      float* __restrict__ out) {
  __shared__ unsigned short lA[2][TILE_CO * A_STRIDE];  // 2 x 34816 B
  __shared__ unsigned short lX[2][X_ROWS * X_STRIDE];   // 2 x 10560 B

  const int tid  = threadIdx.x;
  const int lane = tid & 31;
  const int hf   = lane >> 4;
  const int n16  = lane & 15;
  const int wave = tid >> 5;
  const int wr   = wave >> 1;
  const int wc   = wave & 1;

  const int s0  = blockIdx.x * TILE_S;
  const int co0 = blockIdx.y * TILE_CO;
  const int bz  = blockIdx.z;

  const v8f vzero = {0.f, 0.f, 0.f, 0.f, 0.f, 0.f, 0.f, 0.f};
  v8f acc[2][4];
#pragma unroll
  for (int ti = 0; ti < 2; ++ti)
#pragma unroll
    for (int tj = 0; tj < 4; ++tj) acc[ti][tj] = vzero;

#if HAVE_TDM
  const unsigned ldsA0 = (unsigned)(size_t)(void*)&lA[0][0];
  const unsigned ldsA1 = (unsigned)(size_t)(void*)&lA[1][0];
  const unsigned ldsX0 = (unsigned)(size_t)(void*)&lX[0][0];
  const unsigned ldsX1 = (unsigned)(size_t)(void*)&lX[1][0];

  // prologue: fill buffer 0 with block 0
  if (tid < 32) {
    tdm_issue(Wp, Xp, co0, s0, bz, 0, ldsA0, ldsX0);
    __builtin_amdgcn_s_wait_tensorcnt(0);
  }
  __syncthreads();
#endif

  for (int cb = 0; cb < NCB; ++cb) {
    const int cur = cb & 1;

#if HAVE_TDM
    // issue DMA for the next block into the alternate buffer; it overlaps
    // with this block's WMMA work below.
    if (cb + 1 < NCB && tid < 32) {
      tdm_issue(Wp, Xp, co0, s0, bz, cb + 1,
                cur ? ldsA0 : ldsA1, cur ? ldsX0 : ldsX1);
    }
#else
    // staging fallback: straight b128 copies (no conversions needed)
    {
      const uint4* Wa = (const uint4*)(Wp + ((size_t)co0 * NCB + cb) * 128u);
      for (int i = 0; i < 8; ++i) {
        int idx = i * 256 + tid;          // 2048 uint4 = 128 rows * 16
        int row = idx >> 4;
        int q   = idx & 15;
        uint4 v = Wa[(size_t)row * 1024 + q];  // 8192 ushorts/row = 1024 uint4
        *(uint4*)&lA[cur][row * A_STRIDE + q * 8] = v;
      }
      const uint4* Xa = (const uint4*)(Xp + (((size_t)bz * XP_ROWS + s0) * CIN + cb * CI_BLK));
      for (int i = 0; i < 3; ++i) {
        int idx = i * 256 + tid;          // 528 uint4 = 132 rows * 4
        if (idx < 528) {
          int row = idx >> 2;
          int q   = idx & 3;
          uint4 v = Xa[(size_t)row * (CIN / 8) + q];
          *(uint4*)&lX[cur][row * X_STRIDE + q * 8] = v;
        }
      }
    }
    __syncthreads();
#endif

#pragma unroll
    for (int kk = 0; kk < KW; ++kk) {
      Frag af[2];
#pragma unroll
      for (int ti = 0; ti < 2; ++ti) {
        int row = wr * 32 + ti * 16 + n16;
        int off = row * A_STRIDE + kk * 32 + hf * 8;
        af[ti].u[0] = *(const v4u*)&lA[cur][off];
        af[ti].u[1] = *(const v4u*)&lA[cur][off + 16];
      }
      Frag xf[4];
#pragma unroll
      for (int tj = 0; tj < 4; ++tj) {
        int r   = wc * 64 + tj * 16 + n16 + kk + 1;
        int off = r * X_STRIDE + hf * 8;
        xf[tj].u[0] = *(const v4u*)&lX[cur][off];
        xf[tj].u[1] = *(const v4u*)&lX[cur][off + 16];
      }
#pragma unroll
      for (int ti = 0; ti < 2; ++ti)
#pragma unroll
        for (int tj = 0; tj < 4; ++tj)
          acc[ti][tj] = __builtin_amdgcn_wmma_f32_16x16x32_bf16(
              false, af[ti].bf, false, xf[tj].bf, (short)0, acc[ti][tj],
              false, false);
    }

#if HAVE_TDM
    // drain the in-flight DMA for the next buffer, then flip
    if (tid < 32) __builtin_amdgcn_s_wait_tensorcnt(0);
    __syncthreads();
#else
    __syncthreads();
#endif
  }

#pragma unroll
  for (int ti = 0; ti < 2; ++ti) {
    const int coBase = co0 + wr * 32 + ti * 16;
#pragma unroll
    for (int tj = 0; tj < 4; ++tj) {
      const int sIdx = s0 + wc * 64 + tj * 16 + n16;
#pragma unroll
      for (int j = 0; j < 8; ++j) {
        const int co = coBase + j + hf * 8;
        float v = acc[ti][tj][j] + bias[co];
        float r = v / (1.0f + __expf(-v));
        out[((size_t)bz * COUT + co) * SS + sIdx] = r;
      }
    }
  }
}

// ---------------------------------------------------------------------------
// Fallback kernel: inline conversion staging, no workspace needed
// ---------------------------------------------------------------------------
__global__ __launch_bounds__(256)
void conv_wmma_inline(const float* __restrict__ x,
                      const float* __restrict__ w,
                      const float* __restrict__ bias,
                      float* __restrict__ out) {
  __shared__ unsigned short lA[TILE_CO * A_STRIDE];
  __shared__ unsigned short lX[X_ROWS * X_STRIDE];

  const int tid  = threadIdx.x;
  const int lane = tid & 31;
  const int hf   = lane >> 4;
  const int n16  = lane & 15;
  const int wave = tid >> 5;
  const int wr   = wave >> 1;
  const int wc   = wave & 1;

  const int s0  = blockIdx.x * TILE_S;
  const int co0 = blockIdx.y * TILE_CO;
  const int bz  = blockIdx.z;

  const v8f vzero = {0.f, 0.f, 0.f, 0.f, 0.f, 0.f, 0.f, 0.f};
  v8f acc[2][4];
#pragma unroll
  for (int ti = 0; ti < 2; ++ti)
#pragma unroll
    for (int tj = 0; tj < 4; ++tj) acc[ti][tj] = vzero;

  for (int cb = 0; cb < NCB; ++cb) {
    const int c0 = cb * CI_BLK;

    for (int i = 0; i < 16; ++i) {
      int idx = i * 256 + tid;
      int co  = idx >> 5;
      int ci  = idx & 31;
      const float4 wv = *(const float4*)(w + ((size_t)(co0 + co) * CIN + (c0 + ci)) * KW);
      unsigned short* dst = &lA[co * A_STRIDE + ci];
      dst[0 * 32] = f2bf(wv.x);
      dst[1 * 32] = f2bf(wv.y);
      dst[2 * 32] = f2bf(wv.z);
      dst[3 * 32] = f2bf(wv.w);
    }

    for (int i = 0; i < 5; ++i) {
      int idx = i * 256 + tid;
      if (idx < 32 * 33) {
        int ci = idx / 33;
        int c4 = idx % 33;
        int r0 = c4 * 4;
        int u0 = s0 - 4 + r0;
        const float* xp = x + ((size_t)(bz * CIN + c0 + ci)) * SS;
        float4 v;
        if (u0 >= 0 && u0 + 3 < SS) {
          v = *(const float4*)(xp + u0);
        } else {
          v.x = (u0 + 0 >= 0 && u0 + 0 < SS) ? xp[u0 + 0] : 0.f;
          v.y = (u0 + 1 >= 0 && u0 + 1 < SS) ? xp[u0 + 1] : 0.f;
          v.z = (u0 + 2 >= 0 && u0 + 2 < SS) ? xp[u0 + 2] : 0.f;
          v.w = (u0 + 3 >= 0 && u0 + 3 < SS) ? xp[u0 + 3] : 0.f;
        }
        unsigned short* dst = &lX[r0 * X_STRIDE + ci];
        dst[0 * X_STRIDE] = f2bf(v.x);
        dst[1 * X_STRIDE] = f2bf(v.y);
        dst[2 * X_STRIDE] = f2bf(v.z);
        dst[3 * X_STRIDE] = f2bf(v.w);
      }
    }

    __syncthreads();

#pragma unroll
    for (int kk = 0; kk < KW; ++kk) {
      Frag af[2];
#pragma unroll
      for (int ti = 0; ti < 2; ++ti) {
        int row = wr * 32 + ti * 16 + n16;
        int off = row * A_STRIDE + kk * 32 + hf * 8;
        af[ti].u[0] = *(const v4u*)&lA[off];
        af[ti].u[1] = *(const v4u*)&lA[off + 16];
      }
      Frag xf[4];
#pragma unroll
      for (int tj = 0; tj < 4; ++tj) {
        int r   = wc * 64 + tj * 16 + n16 + kk + 1;
        int off = r * X_STRIDE + hf * 8;
        xf[tj].u[0] = *(const v4u*)&lX[off];
        xf[tj].u[1] = *(const v4u*)&lX[off + 16];
      }
#pragma unroll
      for (int ti = 0; ti < 2; ++ti)
#pragma unroll
        for (int tj = 0; tj < 4; ++tj)
          acc[ti][tj] = __builtin_amdgcn_wmma_f32_16x16x32_bf16(
              false, af[ti].bf, false, xf[tj].bf, (short)0, acc[ti][tj],
              false, false);
    }

    __syncthreads();
  }

#pragma unroll
  for (int ti = 0; ti < 2; ++ti) {
    const int coBase = co0 + wr * 32 + ti * 16;
#pragma unroll
    for (int tj = 0; tj < 4; ++tj) {
      const int sIdx = s0 + wc * 64 + tj * 16 + n16;
#pragma unroll
      for (int j = 0; j < 8; ++j) {
        const int co = coBase + j + hf * 8;
        float v = acc[ti][tj][j] + bias[co];
        float r = v / (1.0f + __expf(-v));
        out[((size_t)bz * COUT + co) * SS + sIdx] = r;
      }
    }
  }
}

// ---------------------------------------------------------------------------
extern "C" void kernel_launch(void* const* d_in, const int* in_sizes, int n_in,
                              void* d_out, int out_size, void* d_ws, size_t ws_size,
                              hipStream_t stream) {
  (void)in_sizes; (void)n_in; (void)out_size;
  const float* x    = (const float*)d_in[0];
  const float* w    = (const float*)d_in[1];
  const float* bias = (const float*)d_in[2];
  float* out        = (float*)d_out;

  dim3 grid(SS / TILE_S, COUT / TILE_CO, CBATCH);  // (32, 16, 4)
  dim3 block(256);

  if (ws_size >= WP_BYTES + XP_BYTES) {
    unsigned short* Wp = (unsigned short*)d_ws;
    unsigned short* Xp = (unsigned short*)((char*)d_ws + WP_BYTES);
    prepack_w<<<dim3((COUT * CIN / 2) / 256), 256, 0, stream>>>(w, Wp);
    prepack_x<<<dim3(SS / 64, CIN / 32, CBATCH), 256, 0, stream>>>(x, Xp);
    zero_xpad<<<dim3(16), 256, 0, stream>>>(Xp);
    conv_wmma_packed<<<grid, block, 0, stream>>>(Wp, Xp, bias, out);
  } else {
    conv_wmma_inline<<<grid, block, 0, stream>>>(x, w, bias, out);
  }
}
